// CausalMultiHeadAttention_80900003987681
// MI455X (gfx1250) — matrix operand order repaired and checked by reading, not compiled, not run on hardware
//
#include <hip/hip_runtime.h>
#include <cstdint>
#include <cstddef>

#define D_MODEL 1024
#define N_HEADS 16
#define DK      64
#define BATCH   2
#define SEQ     2048
#define ROWS    (BATCH * SEQ)   // 4096 total (b,s) rows

typedef __attribute__((ext_vector_type(16))) __bf16 v16bf;
typedef __attribute__((ext_vector_type(8)))  float  v8f;

// ---------------------------------------------------------------------------
// WMMA helper: D = A(16x32 bf16) * B(32x16 bf16) + C(16x16 f32)
// ---------------------------------------------------------------------------
__device__ __forceinline__ v8f wmma_bf16(v16bf a, v16bf b, v8f c) {
  return __builtin_amdgcn_wmma_f32_16x16x32_bf16(
      /*neg_a=*/false, a, /*neg_b=*/false, b,
      /*c_mod=*/(short)0, c, /*reuse_a=*/false, /*reuse_b=*/false);
}

union Frag { uint4 u[2]; v16bf v; };

// A-matrix bf16 16x32 layout (ISA 7.12.2): lane = (g = lane/16, m = lane%16),
// element e -> K = 8*g + e (e<8), 16 + 8*g + (e-8) (e>=8).
// tile points at (row0, k0) of a row-major matrix with row stride ld.
__device__ __forceinline__ v16bf load_a_frag(const __bf16* tile, int ld, int lane) {
  int m = lane & 15, g = lane >> 4;
  const __bf16* p = tile + m * ld + 8 * g;
  Frag f;
  f.u[0] = *(const uint4*)(p);        // K = 8g .. 8g+7
  f.u[1] = *(const uint4*)(p + 16);   // K = 16+8g .. 16+8g+7
  return f.v;
}

// B-matrix bf16 32x16 layout: lane = (g, n = lane%16), element e -> K = 8*g + e (e<8), 16 + 8*g + (e-8) (e>=8).
// Source supplied transposed: B[k][n] = tileT[n*ld + k]  (two runs of 8 contiguous K values).
__device__ __forceinline__ v16bf load_b_fragT(const __bf16* tileT, int ld, int lane) {
  int n = lane & 15, g = lane >> 4;
  const __bf16* p = tileT + n * ld + 8 * g;
  Frag f;
  f.u[0] = *(const uint4*)(p);        // K = 8g .. 8g+7
  f.u[1] = *(const uint4*)(p + 16);   // K = 16+8g .. 16+8g+7
  return f.v;
}

// Cross-lane row reductions inside each 16-lane half (C-layout columns).
__device__ __forceinline__ float rowmax16(float x) {
  x = fmaxf(x, __shfl_xor(x, 8, 32));
  x = fmaxf(x, __shfl_xor(x, 4, 32));
  x = fmaxf(x, __shfl_xor(x, 2, 32));
  x = fmaxf(x, __shfl_xor(x, 1, 32));
  return x;
}
__device__ __forceinline__ float rowsum16(float x) {
  x += __shfl_xor(x, 8, 32);
  x += __shfl_xor(x, 4, 32);
  x += __shfl_xor(x, 2, 32);
  x += __shfl_xor(x, 1, 32);
  return x;
}

// ---------------------------------------------------------------------------
// fp32 -> bf16 conversion (4 elements / thread)
// ---------------------------------------------------------------------------
__global__ __launch_bounds__(256) void convert_f32_bf16(
    const float* __restrict__ src, __bf16* __restrict__ dst, int n4) {
  int i = blockIdx.x * blockDim.x + threadIdx.x;
  if (i < n4) {
    float4 f = ((const float4*)src)[i];
    union { __bf16 h[4]; uint2 u; } o;
    o.h[0] = (__bf16)f.x; o.h[1] = (__bf16)f.y;
    o.h[2] = (__bf16)f.z; o.h[3] = (__bf16)f.w;
    ((uint2*)dst)[i] = o.u;
  }
}

// ---------------------------------------------------------------------------
// QKV projection: y = x * W^T + b. Each wave computes a 32x64 output macro-tile
// (2 M-frags x 4 N-frags = 8 WMMA accumulators): 12 b128 loads per 8 WMMA.
// which = blockIdx.z selects (Wq->Q), (Wk->K), (Wv->V^T).
// Q,K layout [b,h,s,64] bf16; V stored transposed [b,h,64,s] bf16.
// ---------------------------------------------------------------------------
__global__ __launch_bounds__(256) void qkv_gemm(
    const __bf16* __restrict__ xb,
    const __bf16* __restrict__ Wqb, const __bf16* __restrict__ Wkb,
    const __bf16* __restrict__ Wvb,
    const float* __restrict__ bq, const float* __restrict__ bk,
    const float* __restrict__ bv,
    __bf16* __restrict__ Q, __bf16* __restrict__ Kout, __bf16* __restrict__ Vt) {
  int lane  = threadIdx.x & 31;
  int warp  = threadIdx.x >> 5;
  int mbase = blockIdx.x * 32;                  // 128 blocks.x -> rows
  int nbase = (blockIdx.y * 8 + warp) * 64;     // blocks.y = 2 -> cols 0..960
  int which = blockIdx.z;                       // 0,1,2

  const __bf16* W    = (which == 0) ? Wqb : (which == 1) ? Wkb : Wvb;
  const float*  bias = (which == 0) ? bq  : (which == 1) ? bk  : bv;

  int g = lane >> 4, mn = lane & 15;
  const __bf16* A0 = xb + (size_t)(mbase +  0) * D_MODEL;
  const __bf16* A1 = xb + (size_t)(mbase + 16) * D_MODEL;

  v8f acc[2][4] = {};
  for (int k0 = 0; k0 < D_MODEL; k0 += 32) {
    v16bf a0 = load_a_frag(A0 + k0, D_MODEL, lane);
    v16bf a1 = load_a_frag(A1 + k0, D_MODEL, lane);
#pragma unroll
    for (int j = 0; j < 4; ++j) {
      v16bf b = load_b_fragT(W + (size_t)(nbase + 16 * j) * D_MODEL + k0,
                             D_MODEL, lane);
      acc[0][j] = wmma_bf16(a0, b, acc[0][j]);
      acc[1][j] = wmma_bf16(a1, b, acc[1][j]);
    }
  }

#pragma unroll
  for (int j = 0; j < 4; ++j) {
    int   nglob = nbase + 16 * j + mn;          // 0..1023
    float bb    = bias[nglob];
    int   h     = nglob >> 6;
    int   d     = nglob & 63;
#pragma unroll
    for (int i = 0; i < 2; ++i) {
#pragma unroll
      for (int v = 0; v < 8; ++v) {
        int srow = mbase + 16 * i + v + 8 * g;  // 0..4095
        int b_   = srow >> 11;
        int s    = srow & (SEQ - 1);
        __bf16 val = (__bf16)(acc[i][j][v] + bb);
        if (which == 0)
          Q[(size_t)((b_ * N_HEADS + h) * SEQ + s) * DK + d] = val;
        else if (which == 1)
          Kout[(size_t)((b_ * N_HEADS + h) * SEQ + s) * DK + d] = val;
        else
          Vt[(size_t)((b_ * N_HEADS + h) * DK + d) * SEQ + s] = val;
      }
    }
  }
}

// ---------------------------------------------------------------------------
// Flash attention: one wave = one 16-query tile for one (b,h).
// Streams 32-key chunks: 4 WMMA for scores, online softmax, LDS bounce for
// P relayout (C-layout -> A-layout), 4 WMMA for P*V. Next chunk's K/V are
// prefetched (global_prefetch_b8) while the current chunk computes.
// Output bf16 [b,s,1024].
// ---------------------------------------------------------------------------
#define NEG_BIG (-1e30f)

__global__ __launch_bounds__(128) void flash_attn(
    const __bf16* __restrict__ Q, const __bf16* __restrict__ K,
    const __bf16* __restrict__ Vt, __bf16* __restrict__ AO) {
  __shared__ __bf16 lds[4][16][32];       // 1KB per wave P tile

  int lane = threadIdx.x & 31;
  int warp = threadIdx.x >> 5;
  int w    = blockIdx.x * 4 + warp;       // 0..4095
  int t    = w & 127;                     // q tile
  int h    = (w >> 7) & 15;
  int b    = w >> 11;
  int g    = lane >> 4, mn = lane & 15;

  const __bf16* Qh = Q  + (size_t)(b * N_HEADS + h) * SEQ * DK;
  const __bf16* Kh = K  + (size_t)(b * N_HEADS + h) * SEQ * DK;
  const __bf16* Vh = Vt + (size_t)(b * N_HEADS + h) * DK * SEQ;

  v16bf aq0 = load_a_frag(Qh + (size_t)(t * 16) * DK + 0,  DK, lane);
  v16bf aq1 = load_a_frag(Qh + (size_t)(t * 16) * DK + 32, DK, lane);

  v8f o0 = {}, o1 = {}, o2 = {}, o3 = {};
  float m[8], l[8];
#pragma unroll
  for (int v = 0; v < 8; ++v) { m[v] = NEG_BIG; l[v] = 0.f; }

  int nchunk = (t + 2) >> 1;              // ceil(16*(t+1)/32)
  for (int c = 0; c < nchunk; ++c) {
    int kc = c * 32;

    // Prefetch next chunk while this one computes (wave-uniform condition).
    if (c + 1 < nchunk) {
      int kn = kc + 32;
      __builtin_prefetch(Kh + (size_t)(kn + lane) * DK, 0, 3);           // 32 K rows
      __builtin_prefetch(Vh + (size_t)(2 * lane + 0) * SEQ + kn, 0, 3);  // 64 V rows
      __builtin_prefetch(Vh + (size_t)(2 * lane + 1) * SEQ + kn, 0, 3);
    }

    v8f s0 = {}, s1 = {};
    { v16bf bk = load_b_fragT(Kh + (size_t)(kc +  0) * DK + 0,  DK, lane); s0 = wmma_bf16(aq0, bk, s0); }
    { v16bf bk = load_b_fragT(Kh + (size_t)(kc +  0) * DK + 32, DK, lane); s0 = wmma_bf16(aq1, bk, s0); }
    { v16bf bk = load_b_fragT(Kh + (size_t)(kc + 16) * DK + 0,  DK, lane); s1 = wmma_bf16(aq0, bk, s1); }
    { v16bf bk = load_b_fragT(Kh + (size_t)(kc + 16) * DK + 32, DK, lane); s1 = wmma_bf16(aq1, bk, s1); }

    // online softmax (C layout: row = v + 8g, col = mn / 16+mn)
#pragma unroll
    for (int v = 0; v < 8; ++v) {
      int row = t * 16 + v + 8 * g;
      float x0 = (kc + mn      <= row) ? s0[v] * 0.125f : NEG_BIG;
      float x1 = (kc + 16 + mn <= row) ? s1[v] * 0.125f : NEG_BIG;
      float mnew  = fmaxf(m[v], rowmax16(fmaxf(x0, x1)));
      float alpha = __expf(m[v] - mnew);
      float e0 = __expf(x0 - mnew);
      float e1 = __expf(x1 - mnew);
      l[v] = l[v] * alpha + rowsum16(e0 + e1);
      m[v] = mnew;
      o0[v] *= alpha; o1[v] *= alpha; o2[v] *= alpha; o3[v] *= alpha;
      lds[warp][v + 8 * g][mn]      = (__bf16)e0;
      lds[warp][v + 8 * g][16 + mn] = (__bf16)e1;
    }
    asm volatile("s_wait_dscnt 0" ::: "memory");   // LDS in-order; belt&braces

    v16bf pa = load_a_frag(&lds[warp][0][0], 32, lane);
    { v16bf bv = load_b_fragT(Vh + (size_t)( 0) * SEQ + kc, SEQ, lane); o0 = wmma_bf16(pa, bv, o0); }
    { v16bf bv = load_b_fragT(Vh + (size_t)(16) * SEQ + kc, SEQ, lane); o1 = wmma_bf16(pa, bv, o1); }
    { v16bf bv = load_b_fragT(Vh + (size_t)(32) * SEQ + kc, SEQ, lane); o2 = wmma_bf16(pa, bv, o2); }
    { v16bf bv = load_b_fragT(Vh + (size_t)(48) * SEQ + kc, SEQ, lane); o3 = wmma_bf16(pa, bv, o3); }
  }

  float inv[8];
#pragma unroll
  for (int v = 0; v < 8; ++v) inv[v] = 1.0f / l[v];
#pragma unroll
  for (int v = 0; v < 8; ++v) {
    int    srow = b * SEQ + t * 16 + v + 8 * g;
    size_t base = (size_t)srow * D_MODEL + h * DK;
    AO[base +  0 + mn] = (__bf16)(o0[v] * inv[v]);
    AO[base + 16 + mn] = (__bf16)(o1[v] * inv[v]);
    AO[base + 32 + mn] = (__bf16)(o2[v] * inv[v]);
    AO[base + 48 + mn] = (__bf16)(o3[v] * inv[v]);
  }
}

// ---------------------------------------------------------------------------
// Output projection: out = AO * Wo^T + bo (fp32 output), 32x64 per wave.
// ---------------------------------------------------------------------------
__global__ __launch_bounds__(256) void out_proj(
    const __bf16* __restrict__ AO, const __bf16* __restrict__ Wob,
    const float* __restrict__ bo, float* __restrict__ out) {
  int lane  = threadIdx.x & 31;
  int warp  = threadIdx.x >> 5;
  int mbase = blockIdx.x * 32;
  int nbase = (blockIdx.y * 8 + warp) * 64;
  int g = lane >> 4, mn = lane & 15;

  const __bf16* A0 = AO + (size_t)(mbase +  0) * D_MODEL;
  const __bf16* A1 = AO + (size_t)(mbase + 16) * D_MODEL;

  v8f acc[2][4] = {};
  for (int k0 = 0; k0 < D_MODEL; k0 += 32) {
    v16bf a0 = load_a_frag(A0 + k0, D_MODEL, lane);
    v16bf a1 = load_a_frag(A1 + k0, D_MODEL, lane);
#pragma unroll
    for (int j = 0; j < 4; ++j) {
      v16bf b = load_b_fragT(Wob + (size_t)(nbase + 16 * j) * D_MODEL + k0,
                             D_MODEL, lane);
      acc[0][j] = wmma_bf16(a0, b, acc[0][j]);
      acc[1][j] = wmma_bf16(a1, b, acc[1][j]);
    }
  }

#pragma unroll
  for (int j = 0; j < 4; ++j) {
    int   nglob = nbase + 16 * j + mn;
    float bb    = bo[nglob];
#pragma unroll
    for (int i = 0; i < 2; ++i) {
#pragma unroll
      for (int v = 0; v < 8; ++v) {
        int srow = mbase + 16 * i + v + 8 * g;
        out[(size_t)srow * D_MODEL + nglob] = acc[i][j][v] + bb;
      }
    }
  }
}

// ---------------------------------------------------------------------------
// Launch
// ---------------------------------------------------------------------------
extern "C" void kernel_launch(void* const* d_in, const int* in_sizes, int n_in,
                              void* d_out, int out_size, void* d_ws, size_t ws_size,
                              hipStream_t stream) {
  const float* x  = (const float*)d_in[0];
  const float* Wq = (const float*)d_in[1];
  const float* bq = (const float*)d_in[2];
  const float* Wk = (const float*)d_in[3];
  const float* bk = (const float*)d_in[4];
  const float* Wv = (const float*)d_in[5];
  const float* bv = (const float*)d_in[6];
  const float* Wo = (const float*)d_in[7];
  const float* bo = (const float*)d_in[8];

  char* ws = (char*)d_ws;
  const size_t MB = (size_t)1 << 20;
  __bf16* xb  = (__bf16*)(ws +  0 * MB);  // 8 MB  : x bf16 [4096,1024]
  __bf16* Wqb = (__bf16*)(ws +  8 * MB);  // 2 MB
  __bf16* Wkb = (__bf16*)(ws + 10 * MB);  // 2 MB
  __bf16* Wvb = (__bf16*)(ws + 12 * MB);  // 2 MB
  __bf16* Wob = (__bf16*)(ws + 14 * MB);  // 2 MB
  __bf16* Qb  = (__bf16*)(ws + 16 * MB);  // 8 MB  : [b,h,s,64]
  __bf16* Kb  = (__bf16*)(ws + 24 * MB);  // 8 MB  : [b,h,s,64]
  __bf16* Vtb = (__bf16*)(ws + 32 * MB);  // 8 MB  : [b,h,64,s]
  __bf16* AOb = (__bf16*)(ws + 40 * MB);  // 8 MB  : [b,s,1024]

  const int nx4 = (ROWS * D_MODEL) / 4;       // 1,048,576
  const int nw4 = (D_MODEL * D_MODEL) / 4;    //   262,144
  convert_f32_bf16<<<nx4 / 256, 256, 0, stream>>>(x,  xb,  nx4);
  convert_f32_bf16<<<nw4 / 256, 256, 0, stream>>>(Wq, Wqb, nw4);
  convert_f32_bf16<<<nw4 / 256, 256, 0, stream>>>(Wk, Wkb, nw4);
  convert_f32_bf16<<<nw4 / 256, 256, 0, stream>>>(Wv, Wvb, nw4);
  convert_f32_bf16<<<nw4 / 256, 256, 0, stream>>>(Wo, Wob, nw4);

  qkv_gemm<<<dim3(ROWS / 32, 2, 3), 256, 0, stream>>>(
      xb, Wqb, Wkb, Wvb, bq, bk, bv, Qb, Kb, Vtb);

  flash_attn<<<(BATCH * N_HEADS * (SEQ / 16)) / 4, 128, 0, stream>>>(
      Qb, Kb, Vtb, AOb);

  out_proj<<<dim3(ROWS / 32, 2), 256, 0, stream>>>(
      AOb, Wob, bo, (float*)d_out);
}
